// GPAttention_59708635349056
// MI455X (gfx1250) — compile-verified, matrix operand order checked by reading, not run on hardware
//
#include <hip/hip_runtime.h>
#include <hip/hip_bf16.h>

// ---------------------------------------------------------------------------
// GPAttention on MI455X (gfx1250): bf16 WMMA 16x16x32 GEMMs, fp32 epilogues.
// Round 2: software-pipelined GEMM inner loop (1 A-frag feeds 4 independent
// WMMA chains; next k-step fragments prefetched before current WMMAs issue).
// ---------------------------------------------------------------------------

typedef __bf16 bf16;
typedef __attribute__((ext_vector_type(16))) __bf16 v16bf;
typedef __attribute__((ext_vector_type(8)))  __bf16 v8bf;
typedef __attribute__((ext_vector_type(4)))  __bf16 v4bf;
typedef __attribute__((ext_vector_type(2)))  __bf16 v2bf;
typedef __attribute__((ext_vector_type(8)))  float  v8f;
typedef __attribute__((ext_vector_type(4)))  float  v4f;

constexpr int HD = 512;            // hidden dim
constexpr int LL = 1024;           // SES*SEQ
constexpr int NB = 16;             // batch
constexpr int KK = 32;             // keys per query
constexpr int MT = NB * LL;        // 16384 total rows

// ---------------------------------------------------------------------------
// fp32 -> bf16 conversion (vectorized: 4 floats -> 4 bf16 per thread)
// ---------------------------------------------------------------------------
__global__ __launch_bounds__(256) void cvt_bf16(const float* __restrict__ src,
                                                bf16* __restrict__ dst, int n4) {
  int i = blockIdx.x * 256 + threadIdx.x;
  if (i < n4) {
    v4f f = ((const v4f*)src)[i];
    v4bf o = { (bf16)f.x, (bf16)f.y, (bf16)f.z, (bf16)f.w };
    ((v4bf*)dst)[i] = o;
  }
}

// ---------------------------------------------------------------------------
// q = user_emb @ Wq^T + bq   (16 x 512, tiny: plain fp32 dot per thread)
// ---------------------------------------------------------------------------
__global__ __launch_bounds__(256) void qproj(const float* __restrict__ ue,
                                             const float* __restrict__ Wq,
                                             const float* __restrict__ bq,
                                             float* __restrict__ q) {
  int i = blockIdx.x * 256 + threadIdx.x;      // 0 .. NB*HD-1
  int b = i >> 9, n = i & (HD - 1);
  const float* u = ue + b * HD;
  const float* w = Wq + n * HD;
  float s = 0.f;
  for (int j = 0; j < HD; ++j) s = fmaf(u[j], w[j], s);
  q[i] = s + bq[n];
}

// ---------------------------------------------------------------------------
// WMMA fragment load: per-lane 2x 16B chunks (K = [0..7] and [16..23] relative
// to a pointer that already includes the per-lane +half*8 element offset).
// ---------------------------------------------------------------------------
__device__ __forceinline__ v16bf load_frag(const bf16* p) {
  v16bf r;
  *((v8bf*)&r)       = *((const v8bf*)(p));
  *(((v8bf*)&r) + 1) = *((const v8bf*)(p + 16));
  return r;
}

// ---------------------------------------------------------------------------
// Per-wave GEMM strip: 4 N-tiles (16 cols apart by 16 rows of W) accumulated
// over K=512. Software pipelined: k+32 fragments are loaded before the four
// WMMAs of step k issue, so the load clause overlaps matrix work.
//   Arow = LDS A base  + lane16*HD + half*8
//   Wrow = W           + (n_base + lane16)*HD + half*8   (tiles: +16*HD each)
// ---------------------------------------------------------------------------
__device__ __forceinline__ void gemm_strip4(const bf16* __restrict__ Arow,
                                            const bf16* __restrict__ Wrow,
                                            v8f acc[4]) {
  v16bf a  = load_frag(Arow);
  v16bf b0 = load_frag(Wrow);
  v16bf b1 = load_frag(Wrow + 16 * HD);
  v16bf b2 = load_frag(Wrow + 32 * HD);
  v16bf b3 = load_frag(Wrow + 48 * HD);
#pragma unroll
  for (int k0 = 0; k0 < HD; k0 += 32) {
    v16bf an = a, bn0 = b0, bn1 = b1, bn2 = b2, bn3 = b3;
    if (k0 + 32 < HD) {
      an  = load_frag(Arow + k0 + 32);
      bn0 = load_frag(Wrow + k0 + 32);
      bn1 = load_frag(Wrow + 16 * HD + k0 + 32);
      bn2 = load_frag(Wrow + 32 * HD + k0 + 32);
      bn3 = load_frag(Wrow + 48 * HD + k0 + 32);
    }
    acc[0] = __builtin_amdgcn_wmma_f32_16x16x32_bf16(false, a, false, b0,
                                                     (short)0, acc[0], false, false);
    acc[1] = __builtin_amdgcn_wmma_f32_16x16x32_bf16(false, a, false, b1,
                                                     (short)0, acc[1], false, false);
    acc[2] = __builtin_amdgcn_wmma_f32_16x16x32_bf16(false, a, false, b2,
                                                     (short)0, acc[2], false, false);
    acc[3] = __builtin_amdgcn_wmma_f32_16x16x32_bf16(false, a, false, b3,
                                                     (short)0, acc[3], false, false);
    a = an; b0 = bn0; b1 = bn1; b2 = bn2; b3 = bn3;
  }
}

// ---------------------------------------------------------------------------
// K-projection GEMM fused with score computation:
//   k_row = X @ Wk^T + bk   (kept in LDS only, never hits HBM)
//   scores[m] = dot(q[b], k_row) / sqrt(H)
// Block = 256 threads = 8 waves; computes 16 rows x 512 cols.
// ---------------------------------------------------------------------------
__global__ __launch_bounds__(256) void kproj_scores(
    const bf16* __restrict__ Xbf, const bf16* __restrict__ Wkbf,
    const float* __restrict__ bk, const float* __restrict__ q,
    float* __restrict__ scores) {
  __shared__ bf16 As[16 * HD];
  __shared__ float Ks[16 * HD];
  const int tid = threadIdx.x;
  const int m0 = blockIdx.x * 16;

  {  // stage 16x512 bf16 A-strip into LDS (16 KB)
    const v8bf* src = (const v8bf*)(Xbf + (size_t)m0 * HD);
    v8bf* dst = (v8bf*)As;
    for (int i = tid; i < 16 * HD / 8; i += 256) dst[i] = src[i];
  }
  __syncthreads();

  const int wave = tid >> 5, lane = tid & 31;
  const int lane16 = lane & 15, half = lane >> 4;

  v8f acc[4] = {};
  gemm_strip4(As + lane16 * HD + half * 8,
              Wkbf + (size_t)(wave * 64 + lane16) * HD + half * 8, acc);

#pragma unroll
  for (int t = 0; t < 4; ++t) {
    const int N = wave * 64 + t * 16 + lane16;
#pragma unroll
    for (int j = 0; j < 8; ++j) {       // C layout: M = 8*half + j, N = lane16
      Ks[(8 * half + j) * HD + N] = acc[t][j] + bk[N];
    }
  }
  __syncthreads();

  // scores epilogue: 2 rows per wave, wave32 reduction
  const int b = m0 >> 10;
  const float* qb = q + b * HD;
  for (int r = wave; r < 16; r += 8) {
    float s = 0.f;
    for (int h = lane; h < HD; h += 32) s = fmaf(qb[h], Ks[r * HD + h], s);
#pragma unroll
    for (int off = 16; off > 0; off >>= 1) s += __shfl_xor(s, off);
    if (lane == 0) scores[m0 + r] = s * 0.044194173824159216f;  // 1/sqrt(512)
  }
}

// ---------------------------------------------------------------------------
// V-projection GEMM: v = X @ Wv^T + bv, stored bf16 (L2-resident, 16 MB)
// ---------------------------------------------------------------------------
__global__ __launch_bounds__(256) void vproj(
    const bf16* __restrict__ Xbf, const bf16* __restrict__ Wvbf,
    const float* __restrict__ bv, bf16* __restrict__ Vbf) {
  __shared__ bf16 As[16 * HD];
  const int tid = threadIdx.x;
  const int m0 = blockIdx.x * 16;
  {
    const v8bf* src = (const v8bf*)(Xbf + (size_t)m0 * HD);
    v8bf* dst = (v8bf*)As;
    for (int i = tid; i < 16 * HD / 8; i += 256) dst[i] = src[i];
  }
  __syncthreads();

  const int wave = tid >> 5, lane = tid & 31;
  const int lane16 = lane & 15, half = lane >> 4;

  v8f acc[4] = {};
  gemm_strip4(As + lane16 * HD + half * 8,
              Wvbf + (size_t)(wave * 64 + lane16) * HD + half * 8, acc);

#pragma unroll
  for (int t = 0; t < 4; ++t) {
    const int N = wave * 64 + t * 16 + lane16;
#pragma unroll
    for (int j = 0; j < 8; ++j) {
      const int M = 8 * half + j;
      Vbf[(size_t)(m0 + M) * HD + N] = (bf16)(acc[t][j] + bv[N]);
    }
  }
}

// ---------------------------------------------------------------------------
// Gather + softmax (K=32 == wave32: one key per lane) + weighted V sum.
// One wave per (b,l) site; 8 sites per block; v2bf-vectorized V gather.
// ---------------------------------------------------------------------------
__global__ __launch_bounds__(256) void attn(
    const float* __restrict__ scores, const float* __restrict__ mask,
    const int* __restrict__ index, const bf16* __restrict__ Vbf,
    bf16* __restrict__ attn_out) {
  const int wave = threadIdx.x >> 5, lane = threadIdx.x & 31;
  const int site = blockIdx.x * 8 + wave;      // site = b*L + l
  const int b = site >> 10, l = site & (LL - 1);

  const int idx = index[l * KK + lane];
  float s = scores[b * LL + idx] + mask[(size_t)site * KK + lane];

  float mx = s;
#pragma unroll
  for (int off = 16; off > 0; off >>= 1) mx = fmaxf(mx, __shfl_xor(mx, off));
  float e = __expf(s - mx);
  float sum = e;
#pragma unroll
  for (int off = 16; off > 0; off >>= 1) sum += __shfl_xor(sum, off);
  const float w = e / sum;

  const bf16* Vb = Vbf + (size_t)b * LL * HD;
  bf16* out = attn_out + (size_t)site * HD;
  for (int hc = 0; hc < HD; hc += 64) {
    float acc0 = 0.f, acc1 = 0.f;
#pragma unroll
    for (int k = 0; k < KK; ++k) {
      const float wk = __shfl(w, k);
      const int ik = __shfl(idx, k);
      v2bf vv = *(const v2bf*)(Vb + (size_t)ik * HD + hc + 2 * lane);
      acc0 = fmaf(wk, (float)vv.x, acc0);
      acc1 = fmaf(wk, (float)vv.y, acc1);
    }
    v2bf o = { (bf16)acc0, (bf16)acc1 };
    *(v2bf*)(out + hc + 2 * lane) = o;
  }
}

// ---------------------------------------------------------------------------
// Dense GEMM fused with residual add + LayerNorm:
//   x = attn_out @ Wd^T + bd + item_flat ; y = LN(x)*g + b
// ---------------------------------------------------------------------------
__global__ __launch_bounds__(256) void dense_ln(
    const bf16* __restrict__ Attnbf, const bf16* __restrict__ Wdbf,
    const float* __restrict__ bd, const float* __restrict__ item_flat,
    const float* __restrict__ ln_g, const float* __restrict__ ln_b,
    float* __restrict__ out) {
  __shared__ bf16 As[16 * HD];
  __shared__ float Xs[16 * HD];
  const int tid = threadIdx.x;
  const int m0 = blockIdx.x * 16;
  {
    const v8bf* src = (const v8bf*)(Attnbf + (size_t)m0 * HD);
    v8bf* dst = (v8bf*)As;
    for (int i = tid; i < 16 * HD / 8; i += 256) dst[i] = src[i];
  }
  __syncthreads();

  const int wave = tid >> 5, lane = tid & 31;
  const int lane16 = lane & 15, half = lane >> 4;

  v8f acc[4] = {};
  gemm_strip4(As + lane16 * HD + half * 8,
              Wdbf + (size_t)(wave * 64 + lane16) * HD + half * 8, acc);

#pragma unroll
  for (int t = 0; t < 4; ++t) {
    const int N = wave * 64 + t * 16 + lane16;
#pragma unroll
    for (int j = 0; j < 8; ++j) {
      const int M = 8 * half + j;
      Xs[M * HD + N] = acc[t][j] + bd[N] + item_flat[(size_t)(m0 + M) * HD + N];
    }
  }
  __syncthreads();

  // LayerNorm epilogue: 2 rows per wave
  for (int r = wave; r < 16; r += 8) {
    float s1 = 0.f, s2 = 0.f;
    for (int h = lane; h < HD; h += 32) {
      const float x = Xs[r * HD + h];
      s1 += x;
      s2 = fmaf(x, x, s2);
    }
#pragma unroll
    for (int off = 16; off > 0; off >>= 1) {
      s1 += __shfl_xor(s1, off);
      s2 += __shfl_xor(s2, off);
    }
    const float mu = s1 * (1.f / HD);
    const float var = s2 * (1.f / HD) - mu * mu;
    const float inv = rsqrtf(var + 1e-12f);
    float* orow = out + (size_t)(m0 + r) * HD;
    for (int h = lane; h < HD; h += 32) {
      orow[h] = (Xs[r * HD + h] - mu) * inv * ln_g[h] + ln_b[h];
    }
  }
}

// ---------------------------------------------------------------------------
// Launch
// ---------------------------------------------------------------------------
extern "C" void kernel_launch(void* const* d_in, const int* in_sizes, int n_in,
                              void* d_out, int out_size, void* d_ws, size_t ws_size,
                              hipStream_t stream) {
  const float* user_emb = (const float*)d_in[0];
  const float* item_emb = (const float*)d_in[1];   // (B,SES,SEQ,H) == item_flat
  const float* mask     = (const float*)d_in[2];
  const int*   index    = (const int*)d_in[3];
  const float* Wq = (const float*)d_in[4];
  const float* bq = (const float*)d_in[5];
  const float* Wk = (const float*)d_in[6];
  const float* bk = (const float*)d_in[7];
  const float* Wv = (const float*)d_in[8];
  const float* bv = (const float*)d_in[9];
  const float* Wd = (const float*)d_in[10];
  const float* bd = (const float*)d_in[11];
  const float* ln_g = (const float*)d_in[12];
  const float* ln_b = (const float*)d_in[13];
  float* out = (float*)d_out;

  char* ws = (char*)d_ws;
  bf16* Xbf    = (bf16*)(ws);                                 // 16 MB
  bf16* Attnbf = (bf16*)(ws + (size_t)(16u << 20));           // 16 MB
  bf16* Vbf    = (bf16*)(ws + (size_t)(32u << 20));           // 16 MB
  bf16* Wkbf   = (bf16*)(ws + (size_t)(48u << 20));           // 512 KB
  bf16* Wvbf   = (bf16*)(ws + (size_t)(48u << 20) + (512u << 10));
  bf16* Wdbf   = (bf16*)(ws + (size_t)(48u << 20) + (1024u << 10));
  float* q     = (float*)(ws + (size_t)(48u << 20) + (1536u << 10));   // 32 KB
  float* scores = (float*)(ws + (size_t)(48u << 20) + (1536u << 10) + 32768); // 64 KB

  // 1) precision conversion (fp32 -> bf16) for GEMM operands
  {
    const int n4 = MT * HD / 4;                                // 2,097,152
    cvt_bf16<<<(n4 + 255) / 256, 256, 0, stream>>>(item_emb, Xbf, n4);
    const int w4 = HD * HD / 4;                                // 65,536
    cvt_bf16<<<(w4 + 255) / 256, 256, 0, stream>>>(Wk, Wkbf, w4);
    cvt_bf16<<<(w4 + 255) / 256, 256, 0, stream>>>(Wv, Wvbf, w4);
    cvt_bf16<<<(w4 + 255) / 256, 256, 0, stream>>>(Wd, Wdbf, w4);
  }

  // 2) q projection (tiny, fp32)
  qproj<<<(NB * HD) / 256, 256, 0, stream>>>(user_emb, Wq, bq, q);

  // 3) K-proj GEMM fused with scores (k never hits HBM)
  kproj_scores<<<MT / 16, 256, 0, stream>>>(Xbf, Wkbf, bk, q, scores);

  // 4) V-proj GEMM -> bf16 v
  vproj<<<MT / 16, 256, 0, stream>>>(Xbf, Wvbf, bv, Vbf);

  // 5) gather + softmax + weighted sum (wave32 == K=32)
  attn<<<MT / 8, 256, 0, stream>>>(scores, mask, index, Vbf, Attnbf);

  // 6) dense GEMM + residual + LayerNorm -> fp32 output
  dense_ln<<<MT / 16, 256, 0, stream>>>(Attnbf, Wdbf, bd, item_emb, ln_g, ln_b, out);
}
